// MOELoraLinear_63161789055644
// MI455X (gfx1250) — matrix operand order repaired and checked by reading, not compiled, not run on hardware
//
#include <hip/hip_runtime.h>
#include <stdint.h>

// ---------------------------------------------------------------------------
// MoE-LoRA linear for MI455X (gfx1250, wave32, WMMA bf16)
//   out = x @ W0^T + SCALING * sum_e gate[n,e] * (x @ A_e^T) @ B_e^T
// Strategy: fold gate+scaling into h, giving out = x@W0^T + hs@Bcat^T,
// computed with v_wmma_f32_16x16x32_bf16 (f32 accumulate).
// Main GEMM uses double-buffered LDS staging (regs -> LDS ping-pong) so
// global_load latency overlaps the WMMA chain; one barrier per K-step.
// f32->bf16 staging uses v_cvt_pk_bf16_f32 when available.
// ---------------------------------------------------------------------------

#define N_ROWS   8192
#define IN_F     4096
#define OUT_F    4096
#define E_NUM    8
#define R_TOT    128
#define NCAT     144           // 128 h cols + 8 gate logits + 8 zero pad
#define SCALING  0.125f        // 16 / 128

typedef __attribute__((ext_vector_type(16))) __bf16 v16bf;
typedef __attribute__((ext_vector_type(2)))  __bf16 v2bf;
typedef __attribute__((ext_vector_type(8)))  float  v8f;

// ---- f32 -> bf16 round-to-nearest-even ------------------------------------
__device__ __forceinline__ unsigned short f32_to_bf16(float f) {
    union { float f; uint32_t u; } c;
    c.f = f;
    uint32_t u = c.u + (0x7FFFu + ((c.u >> 16) & 1u));
    return (unsigned short)(u >> 16);
}

__device__ __forceinline__ uint32_t pack2_bf16(float lo, float hi) {
#if __has_builtin(__builtin_amdgcn_cvt_pk_bf16_f32)
    union { v2bf b; uint32_t u; } c;
    c.b = __builtin_amdgcn_cvt_pk_bf16_f32(lo, hi);   // v_cvt_pk_bf16_f32
    return c.u;
#else
    return (uint32_t)f32_to_bf16(lo) | ((uint32_t)f32_to_bf16(hi) << 16);
#endif
}

__device__ __forceinline__ uint2 pack4_bf16(float4 v) {
    uint2 r;
    r.x = pack2_bf16(v.x, v.y);
    r.y = pack2_bf16(v.z, v.w);
    return r;
}

// ---- WMMA fragment loads from LDS (bf16, row-major [row][K]) ---------------
// A-matrix 16x32 (MxK): lanes 0-15 -> M=lane, v0..3 K=0..7,  v4..7 K=16..23
//                       lanes16-31 -> M=lane-16, v0..3 K=8..15, v4..7 K=24..31
__device__ __forceinline__ v16bf load_frag_a(const unsigned short* lds,
                                             int row_base, int stride,
                                             int k_base, int lane) {
    const int lm = lane & 15, lh = lane >> 4;
    const unsigned short* p = lds + (row_base + lm) * stride + k_base + lh * 8;
    union { v16bf v; uint4 u[2]; } f;
    f.u[0] = *(const uint4*)(p);        // K = kb + lh*8 .. +7
    f.u[1] = *(const uint4*)(p + 16);   // K = kb + 16 + lh*8 .. +7
    return f.v;
}

// B-matrix 32x16 (KxN): lanes 0-15 -> N=lane,  v0..3 K=0..7,  v4..7 K=16..23
//                       lanes16-31 -> N=lane-16, v0..3 K=8..15, v4..7 K=24..31
// LDS tile stored as [N][K] row-major -> two 8-bf16 runs per lane.
__device__ __forceinline__ v16bf load_frag_b(const unsigned short* lds,
                                             int col_base, int stride,
                                             int k_base, int lane) {
    const int lm = lane & 15, lh = lane >> 4;
    const unsigned short* p = lds + (col_base + lm) * stride + k_base + lh * 8;
    union { v16bf v; uint4 u[2]; } f;
    f.u[0] = *(const uint4*)(p);
    f.u[1] = *(const uint4*)(p + 16);
    return f.v;
}

#define WMMA_BF16(a, b, c) \
    __builtin_amdgcn_wmma_f32_16x16x32_bf16(false, (a), false, (b), (short)0, (c), false, false)

// ---------------------------------------------------------------------------
// Kernel 1: pack weights.
//   Wcat bf16 [144][4096]: rows 0..127 = A (flattened [E*R_PER][IN_F]),
//                          rows 128..135 = Wg, rows 136..143 = 0
//   Bcat bf16 [4096][128]: Bcat[n][e*16+r] = B[e][n][r]
// ---------------------------------------------------------------------------
__global__ __launch_bounds__(256) void moe_prep_kernel(
    const float* __restrict__ Wg, const float* __restrict__ A,
    const float* __restrict__ B,
    unsigned short* __restrict__ Wcat, unsigned short* __restrict__ Bcat) {
    const int idx = blockIdx.x * 256 + threadIdx.x;
    if (idx < NCAT * IN_F) {
        const int j = idx >> 12;        // row 0..143
        const int k = idx & (IN_F - 1);
        float v = 0.0f;
        if (j < 128)      v = A[(size_t)j * IN_F + k];
        else if (j < 136) v = Wg[(size_t)(j - 128) * IN_F + k];
        Wcat[idx] = f32_to_bf16(v);
    }
    const int idx2 = idx - NCAT * IN_F;
    if (idx2 >= 0 && idx2 < OUT_F * R_TOT) {
        const int n = idx2 >> 7;
        const int j = idx2 & 127;
        const int e = j >> 4, r = j & 15;
        Bcat[idx2] = f32_to_bf16(B[((size_t)e * OUT_F + n) * 16 + r]);
    }
}

// ---------------------------------------------------------------------------
// Kernel 2: htmp[8192][144] = x @ Wcat^T   (bf16 WMMA, f32 accumulate)
// 64 blocks of 256 threads; wave w owns rows w*16..w*16+15, all 9 col tiles.
// Dynamic LDS: xs[128][32] bf16 + wc[144][32] bf16 = 17408 bytes.
// ---------------------------------------------------------------------------
__global__ __launch_bounds__(256) void moe_lora_h_gemm(
    const float* __restrict__ x, const unsigned short* __restrict__ Wcat,
    float* __restrict__ htmp) {
    const int tid  = threadIdx.x;
    const int lane = tid & 31;
    const int wave = tid >> 5;                 // 0..7
    const int gm   = blockIdx.x * 128;

    extern __shared__ char smem1[];
    unsigned short* xs = (unsigned short*)smem1;     // [128][32]
    unsigned short* wc = xs + 128 * 32;              // [144][32]

    v8f zero = {};
    v8f acc[9];
    for (int j = 0; j < 9; ++j) acc[j] = zero;

    for (int k0 = 0; k0 < IN_F; k0 += 32) {
        // stage x tile (f32 -> bf16): 128x32 = 1024 float4, 4 per thread
        for (int c = 0; c < 4; ++c) {
            const int idx = tid + (c << 8);
            const int r   = idx >> 3;
            const int col = (idx & 7) << 2;
            float4 v = *(const float4*)(x + (size_t)(gm + r) * IN_F + k0 + col);
            *(uint2*)(xs + r * 32 + col) = pack4_bf16(v);
        }
        // stage Wcat tile (already bf16): 144x32 = 576 uint4
        for (int c = 0; c < 3; ++c) {
            const int idx = tid + (c << 8);
            if (idx < 576) {
                const int r   = idx >> 2;
                const int col = (idx & 3) << 3;
                *(uint4*)(wc + r * 32 + col) =
                    *(const uint4*)(Wcat + (size_t)r * IN_F + k0 + col);
            }
        }
        __syncthreads();
        v16bf a = load_frag_a(xs, wave * 16, 32, 0, lane);
        // batch B-frag loads in groups of 3 so one dscnt wait covers 3 WMMAs
        for (int jj = 0; jj < 9; jj += 3) {
            v16bf b0 = load_frag_b(wc, (jj + 0) * 16, 32, 0, lane);
            v16bf b1 = load_frag_b(wc, (jj + 1) * 16, 32, 0, lane);
            v16bf b2 = load_frag_b(wc, (jj + 2) * 16, 32, 0, lane);
            acc[jj + 0] = WMMA_BF16(a, b0, acc[jj + 0]);
            acc[jj + 1] = WMMA_BF16(a, b1, acc[jj + 1]);
            acc[jj + 2] = WMMA_BF16(a, b2, acc[jj + 2]);
        }
        __syncthreads();
    }

    // epilogue: C/D layout -> row = base + lh*8 + v, col = tile + lm
    const int lm = lane & 15, lh = lane >> 4;
    const int row0 = gm + wave * 16 + lh * 8;
    for (int j = 0; j < 9; ++j) {
        const int col = j * 16 + lm;
        for (int v = 0; v < 8; ++v)
            htmp[(size_t)(row0 + v) * NCAT + col] = acc[j][v];
    }
}

// ---------------------------------------------------------------------------
// Kernel 3: per-row softmax over logits (htmp cols 128..135), fold gate and
// SCALING into h:  hs[n][j] = SCALING * gate[n][j/16] * htmp[n][j]
// ---------------------------------------------------------------------------
__global__ __launch_bounds__(128) void moe_gate_scale(
    const float* __restrict__ htmp, float* __restrict__ hs) {
    const int n = blockIdx.x;
    const int t = threadIdx.x;
    const float* row = htmp + (size_t)n * NCAT;
    float m = row[128];
    for (int e = 1; e < E_NUM; ++e) m = fmaxf(m, row[128 + e]);
    float s = 0.0f;
    for (int e = 0; e < E_NUM; ++e) s += __expf(row[128 + e] - m);
    const int e = t >> 4;
    const float gate = __expf(row[128 + e] - m) / s;
    hs[(size_t)n * R_TOT + t] = SCALING * gate * row[t];
}

// ---------------------------------------------------------------------------
// Kernel 4: out = x @ W0^T + hs @ Bcat^T   (fused, bf16 WMMA)
// Block tile 128x128, 8 waves as 4(M)x2(N), wave tile 32x64 = 2x4 wmma tiles.
// Dynamic LDS 32768 bytes:
//   phase1: ping-pong buffers, each xs[128][32]+ws[128][32] bf16 (16KB x 2)
//   phase2: hls[128][64] + bls[128][64] bf16 (32KB)
// ---------------------------------------------------------------------------
__global__ __launch_bounds__(256) void moe_main_gemm(
    const float* __restrict__ x, const float* __restrict__ W0,
    const float* __restrict__ hs, const unsigned short* __restrict__ Bcat,
    float* __restrict__ out) {
    const int tid  = threadIdx.x;
    const int lane = tid & 31;
    const int wave = tid >> 5;
    const int gm   = blockIdx.y * 128;          // output row block
    const int gn   = blockIdx.x * 128;          // output col block
    const int wm   = (wave >> 1) * 32;          // wave M offset: 0/32/64/96
    const int wn   = (wave & 1) * 64;           // wave N offset: 0/64

    extern __shared__ char smem2[];
    unsigned short* buf0 = (unsigned short*)smem2;  // [xs 4096 | ws 4096] ushort
    unsigned short* buf1 = buf0 + 8192;             // second 16KB buffer

    // per-thread staging slots (constant across K-steps)
    const int sr   = tid >> 3;                  // row 0..31 handled 4x (stride 32)
    const int scol = (tid & 7) << 2;            // f32 col within 32-wide tile

    v8f zero = {};
    v8f acc[2][4];
    for (int i = 0; i < 2; ++i)
        for (int j = 0; j < 4; ++j) acc[i][j] = zero;

    float4 rx[4], rw[4];

    // prologue: fetch K-step 0 into buf0
    for (int c = 0; c < 4; ++c) {
        const int r = sr + c * 32;
        rx[c] = *(const float4*)(x  + (size_t)(gm + r) * IN_F + scol);
        rw[c] = *(const float4*)(W0 + (size_t)(gn + r) * IN_F + scol);
    }
    for (int c = 0; c < 4; ++c) {
        const int r = sr + c * 32;
        *(uint2*)(buf0 + r * 32 + scol)        = pack4_bf16(rx[c]);
        *(uint2*)(buf0 + 4096 + r * 32 + scol) = pack4_bf16(rw[c]);
    }
    __syncthreads();

    // ---- phase 1: K-loop over IN_F, x @ W0^T, double-buffered --------------
    unsigned short* cur = buf0;
    unsigned short* nxt = buf1;
    for (int k0 = 0; k0 < IN_F; k0 += 32) {
        const bool has_next = (k0 + 32) < IN_F;
        if (has_next) {
            // issue next tile's global loads; waited on only at store time
            for (int c = 0; c < 4; ++c) {
                const int r = sr + c * 32;
                rx[c] = *(const float4*)(x  + (size_t)(gm + r) * IN_F + k0 + 32 + scol);
                rw[c] = *(const float4*)(W0 + (size_t)(gn + r) * IN_F + k0 + 32 + scol);
            }
            if (k0 + 64 < IN_F) {
                __builtin_prefetch(x  + (size_t)(gm + sr) * IN_F + k0 + 64 + scol, 0, 1);
                __builtin_prefetch(W0 + (size_t)(gn + sr) * IN_F + k0 + 64 + scol, 0, 1);
            }
        }
        // compute on current buffer: distinct frag regs -> pipelined ds reads
        {
            unsigned short* xs  = cur;
            unsigned short* wsd = cur + 4096;
            v16bf a0 = load_frag_a(xs, wm,      32, 0, lane);
            v16bf a1 = load_frag_a(xs, wm + 16, 32, 0, lane);
            v16bf b0 = load_frag_b(wsd, wn +  0, 32, 0, lane);
            v16bf b1 = load_frag_b(wsd, wn + 16, 32, 0, lane);
            v16bf b2 = load_frag_b(wsd, wn + 32, 32, 0, lane);
            v16bf b3 = load_frag_b(wsd, wn + 48, 32, 0, lane);
            acc[0][0] = WMMA_BF16(a0, b0, acc[0][0]);
            acc[1][0] = WMMA_BF16(a1, b0, acc[1][0]);
            acc[0][1] = WMMA_BF16(a0, b1, acc[0][1]);
            acc[1][1] = WMMA_BF16(a1, b1, acc[1][1]);
            acc[0][2] = WMMA_BF16(a0, b2, acc[0][2]);
            acc[1][2] = WMMA_BF16(a1, b2, acc[1][2]);
            acc[0][3] = WMMA_BF16(a0, b3, acc[0][3]);
            acc[1][3] = WMMA_BF16(a1, b3, acc[1][3]);
        }
        if (has_next) {
            // convert + store into the other buffer (no conflict with readers
            // of `cur`); single barrier per K-step publishes it
            for (int c = 0; c < 4; ++c) {
                const int r = sr + c * 32;
                *(uint2*)(nxt + r * 32 + scol)        = pack4_bf16(rx[c]);
                *(uint2*)(nxt + 4096 + r * 32 + scol) = pack4_bf16(rw[c]);
            }
            __syncthreads();
            unsigned short* t = cur; cur = nxt; nxt = t;
        }
    }
    __syncthreads();   // all reads of last buffer done before phase 2 reuse

    // ---- phase 2: fused LoRA, K=128 over hs @ Bcat^T (two 64-wide halves) --
    unsigned short* hls = (unsigned short*)smem2;   // [128][64]
    unsigned short* bls = hls + 128 * 64;           // [128][64]
    for (int kh = 0; kh < R_TOT; kh += 64) {
        // hs tile f32 -> bf16: 128x64 = 2048 float4, 8 per thread
        for (int c = 0; c < 8; ++c) {
            const int idx = tid + (c << 8);
            const int r   = idx >> 4;
            const int col = (idx & 15) << 2;
            float4 v = *(const float4*)(hs + (size_t)(gm + r) * R_TOT + kh + col);
            *(uint2*)(hls + r * 64 + col) = pack4_bf16(v);
        }
        // Bcat tile (bf16): 128x64 = 1024 uint4, 4 per thread
        for (int c = 0; c < 4; ++c) {
            const int idx = tid + (c << 8);
            const int r   = idx >> 3;
            const int col = (idx & 7) << 3;
            *(uint4*)(bls + r * 64 + col) =
                *(const uint4*)(Bcat + (size_t)(gn + r) * R_TOT + kh + col);
        }
        __syncthreads();
        for (int kk = 0; kk < 64; kk += 32) {
            v16bf a0 = load_frag_a(hls, wm,      64, kk, lane);
            v16bf a1 = load_frag_a(hls, wm + 16, 64, kk, lane);
            v16bf b0 = load_frag_b(bls, wn +  0, 64, kk, lane);
            v16bf b1 = load_frag_b(bls, wn + 16, 64, kk, lane);
            v16bf b2 = load_frag_b(bls, wn + 32, 64, kk, lane);
            v16bf b3 = load_frag_b(bls, wn + 48, 64, kk, lane);
            acc[0][0] = WMMA_BF16(a0, b0, acc[0][0]);
            acc[1][0] = WMMA_BF16(a1, b0, acc[1][0]);
            acc[0][1] = WMMA_BF16(a0, b1, acc[0][1]);
            acc[1][1] = WMMA_BF16(a1, b1, acc[1][1]);
            acc[0][2] = WMMA_BF16(a0, b2, acc[0][2]);
            acc[1][2] = WMMA_BF16(a1, b2, acc[1][2]);
            acc[0][3] = WMMA_BF16(a0, b3, acc[0][3]);
            acc[1][3] = WMMA_BF16(a1, b3, acc[1][3]);
        }
        __syncthreads();
    }

    // ---- epilogue ----------------------------------------------------------
    const int lm = lane & 15, lh = lane >> 4;
    for (int i = 0; i < 2; ++i) {
        const int row0 = gm + wm + i * 16 + lh * 8;
        for (int j = 0; j < 4; ++j) {
            const int col = gn + wn + j * 16 + lm;
            for (int v = 0; v < 8; ++v)
                out[(size_t)(row0 + v) * OUT_F + col] = acc[i][j][v];
        }
    }
}

// ---------------------------------------------------------------------------
// Workspace layout (bytes):
//   Wcat bf16 : [0,        1179648)   144*4096*2
//   Bcat bf16 : [1179648,  2228224)   4096*128*2
//   htmp f32  : [2228224,  6946816)   8192*144*4
//   hs   f32  : [6946816, 11141120)   8192*128*4
// ---------------------------------------------------------------------------
extern "C" void kernel_launch(void* const* d_in, const int* in_sizes, int n_in,
                              void* d_out, int out_size, void* d_ws, size_t ws_size,
                              hipStream_t stream) {
    (void)in_sizes; (void)n_in; (void)out_size; (void)ws_size;
    const float* x  = (const float*)d_in[0];
    const float* W0 = (const float*)d_in[1];
    const float* Wg = (const float*)d_in[2];
    const float* A  = (const float*)d_in[3];
    const float* B  = (const float*)d_in[4];
    float* out = (float*)d_out;

    char* ws = (char*)d_ws;
    unsigned short* Wcat = (unsigned short*)(ws);
    unsigned short* Bcat = (unsigned short*)(ws + 1179648);
    float*          htmp = (float*)(ws + 2228224);
    float*          hs   = (float*)(ws + 6946816);

    // 1) pack weights: (144*4096 + 4096*128) elems / 256 = 4352 blocks
    moe_prep_kernel<<<4352, 256, 0, stream>>>(Wg, A, B, Wcat, Bcat);

    // 2) h + gate logits GEMM: [8192 x 4096] @ [4096 x 144]
    moe_lora_h_gemm<<<N_ROWS / 128, 256, (128 * 32 + 144 * 32) * 2, stream>>>(
        x, Wcat, htmp);

    // 3) softmax gate, fold gate*SCALING into h
    moe_gate_scale<<<N_ROWS, 128, 0, stream>>>(htmp, hs);

    // 4) fused main GEMM: out = x@W0^T + hs@Bcat^T
    dim3 grid2(OUT_F / 128, N_ROWS / 128);
    moe_main_gemm<<<grid2, 256, 32768, stream>>>(x, W0, hs, Bcat, out);
}
